// UnaryModule_26877905339202
// MI455X (gfx1250) — matrix operand order repaired and checked by reading, not compiled, not run on hardware
//
#include <hip/hip_runtime.h>
#include <hip/hip_bf16.h>
#include <math.h>
#include <utility>

// Problem constants (from reference): B=8, P=100, N=50000, D=128, TOPK=512
#define NBATCH 8
#define NP     100
#define NNEG   50000
#define DIM    128
#define TOPK_  512
#define PT     16              // p-rows per workgroup (one WMMA tile of columns)
#define NTILES (NNEG / 16)     // 3125 n-tiles of 16
#define NB1    2048            // histogram bins (11-bit digits; level2 uses 1024)
#define PTILES 7               // ceil(100/16)

typedef _Float16 v16h __attribute__((ext_vector_type(16)));
typedef float    v8f  __attribute__((ext_vector_type(8)));

// Monotonic mapping: float -> uint32 preserving order (larger float -> larger uint)
__device__ __forceinline__ unsigned monokey(float f) {
  unsigned u = __float_as_uint(f);
  return (u & 0x80000000u) ? ~u : (u | 0x80000000u);
}

__device__ __forceinline__ float softplusf(float x) {
  return (x > 20.f) ? x : log1pf(__expf(x));
}

__global__ __launch_bounds__(256)
void topk_bilinear_softmax(const float* __restrict__ fea0,
                           const float* __restrict__ negf,
                           const float* __restrict__ W,
                           const float* __restrict__ biasp,
                           const float* __restrict__ scalep,
                           float* __restrict__ out)
{
  // 128 KB histogram region; after the 3 hist passes it is re-used as:
  //   [0       .. PT*TOPK )  int   selected indices
  //   [PT*TOPK .. 2*PT*TOPK) float bilinear scores
  __shared__ unsigned sBig[PT * NB1];
  __shared__ float    sPosF[PT][DIM];   // pos rows (zero-padded)
  __shared__ float    sPosW[PT][DIM];   // pos @ W
  __shared__ unsigned sPrefix[PT];      // resolved key prefix per row
  __shared__ int      sNeed[PT];        // remaining rank within current bin
  __shared__ int      sSelCnt[PT];
  __shared__ int      sTieCnt[PT];

  const int tid  = threadIdx.x;
  const int lane = tid & 31;
  const int wave = tid >> 5;           // 0..7
  const int half = lane >> 4;          // K sub-block selector (wave32 layout)
  const int sub  = lane & 15;          // row-within-tile / p-column

  const int b    = blockIdx.x / PTILES;
  const int tile = blockIdx.x % PTILES;
  const int p0   = tile * PT;

  const float* negB = negf + (size_t)b * NNEG * DIM;

  // Load 16 pos rows (zero pad rows >= 100)
  for (int i = tid; i < PT * DIM; i += 256) {
    int pr = i >> 7, c = i & 127;
    int p = p0 + pr;
    sPosF[pr][c] = (p < NP) ? fea0[((size_t)b * NP + p) * DIM + c] : 0.0f;
  }
  if (tid < PT) { sPrefix[tid] = 0u; sNeed[tid] = TOPK_; }
  __syncthreads();

  // B fragments = pos^T (128x16), resident in VGPRs for the whole kernel.
  // ISA 16-bit B 32x16 layout: lane holds column `sub`, K = 8*half..+8 and
  // K = 16+8*half..+8 within each 32-step (matches the A fragment order)
  // -> two 8-element chunks of pos row `sub`.
  v16h bfrag[4];
  #pragma unroll
  for (int s = 0; s < 4; ++s) {
    const int ks = s * 32 + 8 * half;
    #pragma unroll
    for (int i = 0; i < 16; ++i)
      bfrag[s][i] = (_Float16)sPosF[sub][ks + (i < 8 ? i : i + 8)];
  }

  // Streaming pass, fully specialized at compile time (no uniform branches in
  // the inner loop). PASS 0..2: histogram a digit of the monotonic key.
  // PASS 3: select (key > cutoff) plus ticketed ties -> exactly 512 indices.
  auto streamPass = [&](auto passC) {
    constexpr int PASS = decltype(passC)::value;
    const unsigned pref  = sPrefix[sub];
    const int      needv = sNeed[sub];
    int* sSel = (int*)sBig;

    for (int nt = wave; nt < NTILES; nt += 8) {
      const int ntBase = nt * 16;
      const float* nrow = negB + (size_t)(ntBase + sub) * DIM;
      if (nt + 8 < NTILES)
        __builtin_prefetch(negB + (size_t)(ntBase + 128 + sub) * DIM, 0, 1);

      // A fragment: 16-bit A 16x32 layout -> lane holds neg row (ntBase+sub),
      // chunks K = ks+8*half..+8 and K = ks+16+8*half..+8 (contiguous fp32).
      v8f acc = {};
      #pragma unroll
      for (int s = 0; s < 4; ++s) {
        const int k0 = s * 32 + 8 * half;
        float4 q0 = *(const float4*)(nrow + k0);
        float4 q1 = *(const float4*)(nrow + k0 + 4);
        float4 q2 = *(const float4*)(nrow + k0 + 16);
        float4 q3 = *(const float4*)(nrow + k0 + 20);
        v16h a;
        a[0]=(_Float16)q0.x;  a[1]=(_Float16)q0.y;  a[2]=(_Float16)q0.z;  a[3]=(_Float16)q0.w;
        a[4]=(_Float16)q1.x;  a[5]=(_Float16)q1.y;  a[6]=(_Float16)q1.z;  a[7]=(_Float16)q1.w;
        a[8]=(_Float16)q2.x;  a[9]=(_Float16)q2.y;  a[10]=(_Float16)q2.z; a[11]=(_Float16)q2.w;
        a[12]=(_Float16)q3.x; a[13]=(_Float16)q3.y; a[14]=(_Float16)q3.z; a[15]=(_Float16)q3.w;
        acc = __builtin_amdgcn_wmma_f32_16x16x32_f16(
                  false, a, false, bfrag[s], (short)0, acc, false, false);
      }

      // D layout: lane = p-column (sub), VGPR r = n-row r + 8*half
      #pragma unroll
      for (int r = 0; r < 8; ++r) {
        const unsigned u = monokey(acc[r]);
        if constexpr (PASS == 0) {
          atomicAdd(&sBig[sub * NB1 + (u >> 21)], 1u);
        } else if constexpr (PASS == 1) {
          if ((u >> 21) == (pref >> 21))
            atomicAdd(&sBig[sub * NB1 + ((u >> 10) & 0x7FFu)], 1u);
        } else if constexpr (PASS == 2) {
          if ((u >> 10) == (pref >> 10))
            atomicAdd(&sBig[sub * NB1 + (u & 0x3FFu)], 1u);
        } else {
          const int n = ntBase + r + 8 * half;
          if (u > pref) {
            int pos = atomicAdd(&sSelCnt[sub], 1);
            if (pos < TOPK_) sSel[sub * TOPK_ + pos] = n;
          } else if (u == pref) {
            int t = atomicAdd(&sTieCnt[sub], 1);
            if (t < needv) {
              int pos = atomicAdd(&sSelCnt[sub], 1);
              if (pos < TOPK_) sSel[sub * TOPK_ + pos] = n;
            }
          }
        }
      }
    }
  };

  // Scan histogram top-down: find the bin containing the current rank.
  auto scanHist = [&](int pass) {
    if (tid < PT) {
      const int shift = (pass == 0) ? 21 : (pass == 1 ? 10 : 0);
      const int nb    = (pass == 2) ? 1024 : 2048;
      int target = sNeed[tid];
      int cum = 0, c = 0;
      for (int bin = nb - 1; bin >= 0; --bin) {
        int cnt = (int)sBig[tid * NB1 + bin];
        if (cum + cnt >= target) { c = bin; break; }
        cum += cnt;
      }
      sPrefix[tid] |= ((unsigned)c << shift);
      sNeed[tid] = target - cum;   // rank within bin c (>=1)
    }
  };

  auto clearHist = [&]() {
    for (int i = tid; i < PT * NB1; i += 256) sBig[i] = 0u;
  };

  // ---- Passes 0..2: radix-refine the exact 512th-largest key (11+11+10 bits)
  clearHist(); __syncthreads();
  streamPass(std::integral_constant<int, 0>{}); __syncthreads();
  scanHist(0); __syncthreads();

  clearHist(); __syncthreads();
  streamPass(std::integral_constant<int, 1>{}); __syncthreads();
  scanHist(1); __syncthreads();

  clearHist(); __syncthreads();
  streamPass(std::integral_constant<int, 2>{}); __syncthreads();
  scanHist(2); __syncthreads();

  // ---- Pass 3: selection
  if (tid < PT) { sSelCnt[tid] = 0; sTieCnt[tid] = 0; }
  __syncthreads();
  streamPass(std::integral_constant<int, 3>{});
  __syncthreads();

  // ---- Final phase: posW = pos @ W  (W row-major [d][e])
  for (int i = tid; i < PT * DIM; i += 256) {
    int pr = i >> 7, e = i & 127;
    float acc = 0.f;
    for (int d = 0; d < DIM; ++d)
      acc = fmaf(sPosF[pr][d], W[d * DIM + e], acc);
    sPosW[pr][e] = acc;
  }
  __syncthreads();

  // Bilinear scores for the 512 selected negatives per row (gathers hit L2)
  int*   sSel   = (int*)sBig;
  float* sScore = (float*)sBig + PT * TOPK_;
  const float bias = biasp[0];
  for (int i = tid; i < PT * TOPK_; i += 256) {
    const int pr = i >> 9, k = i & (TOPK_ - 1);
    unsigned idx = (unsigned)sSel[pr * TOPK_ + k];
    if (idx >= NNEG) idx = 0;  // safety clamp
    const float* nv = negB + (size_t)idx * DIM;
    float acc = 0.f;
    #pragma unroll 2
    for (int d = 0; d < DIM; d += 4) {
      float4 q = *(const float4*)(nv + d);
      acc = fmaf(sPosW[pr][d + 0], q.x, acc);
      acc = fmaf(sPosW[pr][d + 1], q.y, acc);
      acc = fmaf(sPosW[pr][d + 2], q.z, acc);
      acc = fmaf(sPosW[pr][d + 3], q.w, acc);
    }
    sScore[pr * TOPK_ + k] = acc + bias;
  }
  __syncthreads();

  // Softmax-weighted sum over K=512; one wave32 per row (2 rows per wave)
  const float scl = softplusf(scalep[0]);
  for (int pr = wave; pr < PT; pr += 8) {
    float m = -3.4e38f;
    for (int k = lane; k < TOPK_; k += 32)
      m = fmaxf(m, scl * sScore[pr * TOPK_ + k]);
    #pragma unroll
    for (int off = 16; off > 0; off >>= 1)
      m = fmaxf(m, __shfl_xor(m, off, 32));

    float z = 0.f, ws = 0.f;
    for (int k = lane; k < TOPK_; k += 32) {
      float sc = sScore[pr * TOPK_ + k];
      float e = __expf(scl * sc - m);
      z += e; ws += e * sc;
    }
    #pragma unroll
    for (int off = 16; off > 0; off >>= 1) {
      z  += __shfl_xor(z,  off, 32);
      ws += __shfl_xor(ws, off, 32);
    }
    if (lane == 0 && (p0 + pr) < NP)
      out[b * NP + p0 + pr] = ws / z;
  }
}

extern "C" void kernel_launch(void* const* d_in, const int* in_sizes, int n_in,
                              void* d_out, int out_size, void* d_ws, size_t ws_size,
                              hipStream_t stream) {
  (void)in_sizes; (void)n_in; (void)out_size; (void)d_ws; (void)ws_size;
  const float* fea0 = (const float*)d_in[0];
  const float* negf = (const float*)d_in[1];
  const float* W    = (const float*)d_in[2];
  const float* bias = (const float*)d_in[3];
  const float* scl  = (const float*)d_in[4];
  float* out = (float*)d_out;

  dim3 grid(NBATCH * PTILES);   // 56 workgroups: 8 batches x 7 p-tiles
  dim3 block(256);              // 8 wave32 waves
  topk_bilinear_softmax<<<grid, block, 0, stream>>>(fea0, negf, W, bias, scl, out);
}
